// MemoryEfficientSelectiveScan_77378130804842
// MI455X (gfx1250) — matrix operand order repaired and checked by reading, not compiled, not run on hardware
//
#include <hip/hip_runtime.h>
#include <hip/hip_bf16.h>

// Problem constants (match reference)
#define BB 4
#define TT 4096
#define DD 1024
#define NN 16
#define RR 64

typedef __attribute__((ext_vector_type(16))) __bf16 v16bf;
typedef __attribute__((ext_vector_type(8)))  float  v8f;

union ABFrag { v16bf v; uint4 u[2]; };

// fp32 -> bf16, round to nearest even
__device__ __forceinline__ unsigned short f2bf(float f) {
  unsigned u = __builtin_bit_cast(unsigned, f);
  u += 0x7FFFu + ((u >> 16) & 1u);
  return (unsigned short)(u >> 16);
}

// LDS byte offset of a generic pointer to __shared__ memory
typedef __attribute__((address_space(3))) unsigned char lds_uchar;
__device__ __forceinline__ unsigned lds_off(void* p) {
  return (unsigned)(unsigned long long)(lds_uchar*)p;
}

// Async DMA: 16 bytes global -> LDS, tracked by ASYNCcnt (per-lane addresses).
__device__ __forceinline__ void async_ld_b128(void* lds_dst, const void* gsrc) {
  unsigned laddr = lds_off(lds_dst);
  unsigned long long gaddr = (unsigned long long)gsrc;
  asm volatile("global_load_async_to_lds_b128 %0, %1, off"
               :: "v"(laddr), "v"(gaddr) : "memory");
}

// A-fragment (16x32 bf16, row-major source, leading dim ld):
// lanes 0-15 hold row M=lane, K = k0+0..7 and k0+16..23
// lanes 16-31 hold row M=lane-16, K = k0+8..15 and k0+24..31
__device__ __forceinline__ v16bf loadA(const unsigned short* base, int ld,
                                       int k0, int mrow, int hi16) {
  ABFrag a;
  const unsigned short* p = base + mrow * ld + k0 + hi16 * 8;
  a.u[0] = *(const uint4*)(p);
  a.u[1] = *(const uint4*)(p + 16);
  return a.v;
}

// B-fragment (32x16 bf16) built from row-major weight W[nrows x ld] where
// B[k][n] = W[n][k]: lanes 0-15 hold col N=lane, K = k0+0..7 and k0+16..23;
// lanes 16-31 hold col N=lane-16, K = k0+8..15 and k0+24..31.
__device__ __forceinline__ v16bf loadB(const unsigned short* W, int ld,
                                       int k0, int nrow0, int mrow, int hi16) {
  ABFrag b;
  const unsigned short* p = W + (size_t)(nrow0 + mrow) * ld + k0 + hi16 * 8;
  b.u[0] = *(const uint4*)(p);
  b.u[1] = *(const uint4*)(p + 16);
  return b.v;
}

__global__ void cvt_bf16(const float* __restrict__ src,
                         unsigned short* __restrict__ dst, int n) {
  int i = blockIdx.x * blockDim.x + threadIdx.x;
  if (i < n) dst[i] = f2bf(src[i]);
}

// One wave per 16-row tile of [B*T, D]. Computes Bt, Ct tiles (K=1024 sweep),
// xr = x@W_dt1^T (4 col tiles), then dt = softplus(xr@W_dt2^T + b) fused.
__global__ __launch_bounds__(32)
void proj_kernel(const float* __restrict__ x,
                 const unsigned short* __restrict__ WbB,
                 const unsigned short* __restrict__ WcB,
                 const unsigned short* __restrict__ W1B,
                 const unsigned short* __restrict__ W2B,
                 const float* __restrict__ bdt2,
                 float* __restrict__ BtW, float* __restrict__ CtW,
                 float* __restrict__ dtW) {
  __shared__ __align__(16) unsigned short xs[16 * DD];   // 32 KB: x tile bf16
  __shared__ __align__(16) unsigned short xrs[16 * RR];  // 2 KB:  xr tile bf16

  const int lane = threadIdx.x;
  const int mrow = lane & 15;
  const int hi16 = lane >> 4;
  const int row0 = blockIdx.x * 16;        // tile base row in [0, B*T)

  // Stage x tile fp32 -> bf16 LDS (tile rows are contiguous in [B*T, D])
  const float* xt = x + (size_t)row0 * DD;
  for (int it = 0; it < (16 * DD) / (32 * 4); ++it) {
    int i = (it * 32 + lane) * 4;
    float4 f = *(const float4*)(xt + i);
    unsigned lo = (unsigned)f2bf(f.x) | ((unsigned)f2bf(f.y) << 16);
    unsigned hi = (unsigned)f2bf(f.z) | ((unsigned)f2bf(f.w) << 16);
    *(uint2*)(&xs[i]) = make_uint2(lo, hi);
  }
  __syncthreads();

  // ---- Bt tile: [16 x 16], K sweep over D ----
  {
    v8f acc = {0.f, 0.f, 0.f, 0.f, 0.f, 0.f, 0.f, 0.f};
    for (int k0 = 0; k0 < DD; k0 += 32) {
      v16bf a = loadA(xs, DD, k0, mrow, hi16);
      v16bf b = loadB(WbB, DD, k0, 0, mrow, hi16);
      acc = __builtin_amdgcn_wmma_f32_16x16x32_bf16(false, a, false, b,
                                                    (short)0, acc, false, false);
    }
    float* o = BtW + (size_t)row0 * NN;
#pragma unroll
    for (int r = 0; r < 8; ++r) o[(size_t)(r + 8 * hi16) * NN + mrow] = acc[r];
  }

  // ---- Ct tile ----
  {
    v8f acc = {0.f, 0.f, 0.f, 0.f, 0.f, 0.f, 0.f, 0.f};
    for (int k0 = 0; k0 < DD; k0 += 32) {
      v16bf a = loadA(xs, DD, k0, mrow, hi16);
      v16bf b = loadB(WcB, DD, k0, 0, mrow, hi16);
      acc = __builtin_amdgcn_wmma_f32_16x16x32_bf16(false, a, false, b,
                                                    (short)0, acc, false, false);
    }
    float* o = CtW + (size_t)row0 * NN;
#pragma unroll
    for (int r = 0; r < 8; ++r) o[(size_t)(r + 8 * hi16) * NN + mrow] = acc[r];
  }

  // ---- xr = x @ W_dt1^T : [16 x 64], 4 column tiles ----
  {
    v8f acc0 = {0.f,0.f,0.f,0.f,0.f,0.f,0.f,0.f};
    v8f acc1 = acc0, acc2 = acc0, acc3 = acc0;
    for (int k0 = 0; k0 < DD; k0 += 32) {
      v16bf a = loadA(xs, DD, k0, mrow, hi16);
      v16bf b0 = loadB(W1B, DD, k0, 0,  mrow, hi16);
      v16bf b1 = loadB(W1B, DD, k0, 16, mrow, hi16);
      v16bf b2 = loadB(W1B, DD, k0, 32, mrow, hi16);
      v16bf b3 = loadB(W1B, DD, k0, 48, mrow, hi16);
      acc0 = __builtin_amdgcn_wmma_f32_16x16x32_bf16(false, a, false, b0, (short)0, acc0, false, false);
      acc1 = __builtin_amdgcn_wmma_f32_16x16x32_bf16(false, a, false, b1, (short)0, acc1, false, false);
      acc2 = __builtin_amdgcn_wmma_f32_16x16x32_bf16(false, a, false, b2, (short)0, acc2, false, false);
      acc3 = __builtin_amdgcn_wmma_f32_16x16x32_bf16(false, a, false, b3, (short)0, acc3, false, false);
    }
    // Re-layout D-fragment (lane=N) into row-major bf16 LDS for next GEMM's A
#pragma unroll
    for (int r = 0; r < 8; ++r) {
      int m = r + 8 * hi16;
      xrs[m * RR + 0  + mrow] = f2bf(acc0[r]);
      xrs[m * RR + 16 + mrow] = f2bf(acc1[r]);
      xrs[m * RR + 32 + mrow] = f2bf(acc2[r]);
      xrs[m * RR + 48 + mrow] = f2bf(acc3[r]);
    }
  }
  __syncthreads();

  // ---- dt = softplus(xr @ W_dt2^T + b_dt2) : 64 column tiles, K=64 ----
  for (int c = 0; c < DD / 16; ++c) {
    v8f acc = {0.f, 0.f, 0.f, 0.f, 0.f, 0.f, 0.f, 0.f};
#pragma unroll
    for (int k0 = 0; k0 < RR; k0 += 32) {
      v16bf a = loadA(xrs, RR, k0, mrow, hi16);
      v16bf b = loadB(W2B, RR, k0, c * 16, mrow, hi16);
      acc = __builtin_amdgcn_wmma_f32_16x16x32_bf16(false, a, false, b,
                                                    (short)0, acc, false, false);
    }
    int dcol = c * 16 + mrow;
    float bv = bdt2[dcol];
    float* o = dtW + (size_t)row0 * DD + dcol;
#pragma unroll
    for (int r = 0; r < 8; ++r) {
      float z = acc[r] + bv;
      float sp = (z > 20.0f) ? z : log1pf(__expf(z));  // softplus
      o[(size_t)(r + 8 * hi16) * DD] = sp;
    }
  }
}

// One lane per (b, d, n): 65536 threads = 2048 wave32s. Block = 256 threads
// covers 16 channels of one batch. Chunks of 64 timesteps are double-buffered
// in LDS via async global->LDS DMA (ASYNCcnt); y-reduction over n via shfl_xor.
__global__ __launch_bounds__(256)
void scan_kernel(const float* __restrict__ x, const float* __restrict__ state0,
                 const float* __restrict__ logA, const float* __restrict__ Dsk,
                 const float* __restrict__ BtW, const float* __restrict__ CtW,
                 const float* __restrict__ dtW, float* __restrict__ y) {
  const int CH = 64;
  // [buffer][array][CH*16]: arrays are x, dt, Bt, Ct
  __shared__ __align__(16) float stage[2][4 * CH * 16];   // 32 KB

  const int tid = threadIdx.x;
  const int n  = tid & 15;
  const int ch = tid >> 4;                   // 0..15 channel within block
  const int blocksPerB = DD / 16;            // 64
  const int b  = blockIdx.x / blocksPerB;
  const int d0 = (blockIdx.x % blocksPerB) * 16;
  const int d  = d0 + ch;

  const float A   = -__expf(logA[d * NN + n]);
  const float rA  = 1.0f / (A + 1e-8f);
  float s = state0[((size_t)b * DD + d) * NN + n];
  const float dskip = Dsk[d];

  const size_t xbase  = (size_t)b * TT * DD;
  const size_t bcbase = (size_t)b * TT * NN;

  // each thread DMAs 4 contiguous floats per array (16B, aligned both sides)
  const int i4 = tid * 4;                    // 0..1020
  const int stt = i4 >> 4, sdd = i4 & 15;    // element coords for x/dt arrays

  auto issue_chunk = [&](int t0, int bufsel) {
    float* buf = stage[bufsel];
    const float* gx = x   + xbase + (size_t)(t0 + stt) * DD + d0 + sdd;
    const float* gd = dtW + xbase + (size_t)(t0 + stt) * DD + d0 + sdd;
    const float* gb = BtW + bcbase + (size_t)t0 * NN + i4;
    const float* gc = CtW + bcbase + (size_t)t0 * NN + i4;
    async_ld_b128(&buf[0 * 1024 + i4], gx);
    async_ld_b128(&buf[1 * 1024 + i4], gd);
    async_ld_b128(&buf[2 * 1024 + i4], gb);
    async_ld_b128(&buf[3 * 1024 + i4], gc);
  };

  issue_chunk(0, 0);                          // preload first chunk
  int buf = 0;

  for (int t0 = 0; t0 < TT; t0 += CH) {
    if (t0 + CH < TT) {
      issue_chunk(t0 + CH, buf ^ 1);          // overlap DMA of next chunk
      // loads complete in order: current buffer's 4 done once only the
      // newly-issued 4 remain outstanding
      asm volatile("s_wait_asynccnt 4" ::: "memory");
    } else {
      asm volatile("s_wait_asynccnt 0" ::: "memory");
    }
    __syncthreads();                          // all waves' DMA landed

    const float* xsh = &stage[buf][0 * 1024];
    const float* dsh = &stage[buf][1 * 1024];
    const float* bsh = &stage[buf][2 * 1024];
    const float* csh = &stage[buf][3 * 1024];

    for (int tt = 0; tt < CH; ++tt) {
      float xv = xsh[tt * 16 + ch];
      float dv = dsh[tt * 16 + ch];
      float bn = bsh[tt * 16 + n];
      float cn = csh[tt * 16 + n];
      float abar = __expf(dv * A);                       // in (0,1]
      s = abar * s + ((abar - 1.0f) * rA) * xv * bn;     // ZOH discretization
      float yp = s * cn;
      yp += __shfl_xor(yp, 1, 32);
      yp += __shfl_xor(yp, 2, 32);
      yp += __shfl_xor(yp, 4, 32);
      yp += __shfl_xor(yp, 8, 32);                       // sum over n (16 lanes)
      if (n == 0) y[xbase + (size_t)(t0 + tt) * DD + d] = yp + dskip * xv;
    }
    __syncthreads();   // everyone done reading buf before it's DMA'd again
    buf ^= 1;
  }
}

extern "C" void kernel_launch(void* const* d_in, const int* in_sizes, int n_in,
                              void* d_out, int out_size, void* d_ws, size_t ws_size,
                              hipStream_t stream) {
  const float* x      = (const float*)d_in[0];
  const float* state  = (const float*)d_in[1];
  const float* log_A  = (const float*)d_in[2];
  const float* W_B    = (const float*)d_in[3];
  const float* W_C    = (const float*)d_in[4];
  const float* W_dt1  = (const float*)d_in[5];
  const float* W_dt2  = (const float*)d_in[6];
  const float* b_dt2  = (const float*)d_in[7];
  const float* D_skip = (const float*)d_in[8];
  float* out = (float*)d_out;

  // workspace layout (bytes)
  char* ws = (char*)d_ws;
  unsigned short* WbB = (unsigned short*)(ws + 0);        // 16*1024 bf16
  unsigned short* WcB = (unsigned short*)(ws + 32768);    // 16*1024 bf16
  unsigned short* W1B = (unsigned short*)(ws + 65536);    // 64*1024 bf16
  unsigned short* W2B = (unsigned short*)(ws + 196608);   // 1024*64 bf16
  float* BtW = (float*)(ws + 327680);                     // B*T*N f32 (1 MB)
  float* CtW = (float*)(ws + 1376256);                    // B*T*N f32 (1 MB)
  float* dtW = (float*)(ws + 2424832);                    // B*T*D f32 (64 MB)

  cvt_bf16<<<(16 * 1024 + 255) / 256, 256, 0, stream>>>(W_B,   WbB, 16 * 1024);
  cvt_bf16<<<(16 * 1024 + 255) / 256, 256, 0, stream>>>(W_C,   WcB, 16 * 1024);
  cvt_bf16<<<(64 * 1024 + 255) / 256, 256, 0, stream>>>(W_dt1, W1B, 64 * 1024);
  cvt_bf16<<<(64 * 1024 + 255) / 256, 256, 0, stream>>>(W_dt2, W2B, 64 * 1024);

  proj_kernel<<<(BB * TT) / 16, 32, 0, stream>>>(x, WbB, WcB, W1B, W2B, b_dt2,
                                                 BtW, CtW, dtW);

  scan_kernel<<<BB * (DD / 16), 256, 0, stream>>>(x, state, log_A, D_skip,
                                                  BtW, CtW, dtW, out);
}
